// Attention_60395830116361
// MI455X (gfx1250) — compile-verified, matrix operand order checked by reading, not run on hardware
//
#include <hip/hip_runtime.h>
#include <hip/hip_bf16.h>

// MHA forward for MI455X (gfx1250, wave32).
// Pipeline: 3x projection GEMM (fp32->bf16, WMMA bf16, f32 accum; Q pre-scaled
//           by 1/sqrt(hd)*log2(e), V stored head-transposed)
//           -> flash attention (WMMA bf16, online softmax in exp2 domain,
//              K/V tiles staged via GLOBAL_LOAD_ASYNC_TO_LDS_B128 inline asm)
//           -> output projection GEMM (+bias, fp32 out).
// Workspace (bf16): Q | K (head-split [b,h,t,64]) | V ([b,h,64,t]) | attn-out [b,t,d]
// = 4 * 4,194,304 elems * 2B = 32 MiB (whole working set fits in 192 MB L2).

#define B_  2
#define T_  2048
#define D_  1024
#define H_  16
#define HD_ 64
#define BT_ (B_ * T_)

typedef __bf16 bf16;
typedef __attribute__((ext_vector_type(16))) __bf16        v16bf;
typedef __attribute__((ext_vector_type(8)))  float         v8f;
typedef __attribute__((ext_vector_type(8)))  unsigned int  v8u;

// Async global->LDS copy (ASYNCcnt-tracked, no VGPR round trip).
// ISA 10: "GLOBAL_LOAD_ASYNC_TO_LDS_B128  vdst(lds addr), vaddr(64b), off".
// Generic pointers to __shared__ data carry the LDS byte offset in their low
// 32 bits (ISA 10.2 aperture rules), so truncation yields the LDS address.
__device__ __forceinline__ void async_b128(const bf16* g, bf16* l) {
    unsigned lds = (unsigned)(unsigned long long)l;
    asm volatile("global_load_async_to_lds_b128 %0, %1, off"
                 :: "v"(lds), "v"(g) : "memory");
}

__device__ __forceinline__ void wait_async() {
    asm volatile("s_wait_asynccnt 0" ::: "memory");
}

__device__ __forceinline__ v8f v8f_zero() {
    v8f z;
    #pragma unroll
    for (int i = 0; i < 8; ++i) z[i] = 0.0f;
    return z;
}

// K index inside a 16-bit A/B fragment (ISA 7.12.2):
// lanes 0-15: VGPR0..3 -> K=0..7 pairs, VGPR4..7 -> K=16..23 pairs; lanes 16-31: +8.
__device__ __forceinline__ int frag_k(int v, int lane) {
    int kk0 = (lane & 16) ? 8 : 0;
    return ((v < 4) ? (2 * v) : (16 + 2 * (v - 4))) + kk0;
}

// Load one lane's slice of a 16x32 A-frag / 32x16 B-frag from a row-major tile.
__device__ __forceinline__ v16bf load_frag_row(const bf16* rowp, int lane) {
    union { v16bf b; v8u u; } f;
    #pragma unroll
    for (int v = 0; v < 8; ++v)
        f.u[v] = *(const unsigned int*)(rowp + frag_k(v, lane));
    return f.b;
}

// ---------------------------------------------------------------------------
// Tiled GEMM: out = A(4096x1024) * W^T(1024x1024) [+ bias]
// Block: 128 thr (4 waves), 64x64 tile; wave: 16 rows x 64 cols; K step 32.
// OUTMODE: 0 = head-split bf16 [b,h,t,64] (scaled by oscale)
//          1 = head-transposed bf16 [b,h,64,t]
//          2 = flat fp32 [bt,d] + bias
// ---------------------------------------------------------------------------
template<bool A_BF16, int OUTMODE>
__global__ __launch_bounds__(128, 2)
void gemm_wmma_kernel(const void* __restrict__ Av, const float* __restrict__ W,
                      const float* __restrict__ bias, void* __restrict__ outv,
                      float oscale)
{
    __shared__ bf16 As[64 * 36];   // [row][k], pad 32->36 vs bank conflicts
    __shared__ bf16 Bs[64 * 36];   // [outcol j][k]

    const int tid  = threadIdx.x;
    const int lane = tid & 31;
    const int wid  = tid >> 5;
    const int row0 = blockIdx.y * 64;
    const int col0 = blockIdx.x * 64;

    const float* Af = (const float*)Av;
    const bf16*  Ab = (const bf16*)Av;

    v8f c[4];
    #pragma unroll
    for (int s = 0; s < 4; ++s) c[s] = v8f_zero();

    for (int k0 = 0; k0 < D_; k0 += 32) {
        __syncthreads();
        #pragma unroll
        for (int i = tid; i < 64 * 32; i += 128) {
            int rr = i >> 5, cc = i & 31;
            float a;
            if (A_BF16) a = (float)Ab[(size_t)(row0 + rr) * D_ + k0 + cc];
            else        a = Af[(size_t)(row0 + rr) * D_ + k0 + cc];
            As[rr * 36 + cc] = (bf16)a;
            Bs[rr * 36 + cc] = (bf16)W[(size_t)(col0 + rr) * D_ + k0 + cc];
        }
        __syncthreads();

        if (k0 + 32 < D_) {  // pull next W tile toward L2 (global_prefetch_b8)
            __builtin_prefetch(&W[(size_t)(col0 + (tid >> 1)) * D_ + k0 + 32], 0, 1);
        }

        const bf16* arow = &As[(wid * 16 + (lane & 15)) * 36];
        v16bf af = load_frag_row(arow, lane);
        #pragma unroll
        for (int sub = 0; sub < 4; ++sub) {
            const bf16* brow = &Bs[(sub * 16 + (lane & 15)) * 36];
            v16bf bfr = load_frag_row(brow, lane);
            c[sub] = __builtin_amdgcn_wmma_f32_16x16x32_bf16(
                false, af, false, bfr, (short)0, c[sub], false, false);
        }
    }

    // C layout (ISA 7.12.2): VGPR r -> M=r (lanes 0-15) / M=8+r (lanes 16-31),
    // N = lane & 15.
    #pragma unroll
    for (int sub = 0; sub < 4; ++sub) {
        #pragma unroll
        for (int r = 0; r < 8; ++r) {
            int m   = wid * 16 + r + ((lane & 16) ? 8 : 0);
            int row = row0 + m;
            int j   = col0 + sub * 16 + (lane & 15);
            float val = c[sub][r];
            if (OUTMODE == 0) {
                int b = row >> 11, t = row & (T_ - 1);
                int h = j >> 6,   n = j & 63;
                ((bf16*)outv)[((size_t)(b * H_ + h) * T_ + t) * HD_ + n] =
                    (bf16)(val * oscale);
            } else if (OUTMODE == 1) {
                int b = row >> 11, t = row & (T_ - 1);
                int h = j >> 6,   n = j & 63;
                ((bf16*)outv)[((size_t)(b * H_ + h) * HD_ + n) * T_ + t] = (bf16)val;
            } else {
                ((float*)outv)[(size_t)row * D_ + j] = val + bias[j];
            }
        }
    }
}

// ---------------------------------------------------------------------------
// Flash attention. Q/K: bf16 [bh][t][64] (Q pre-scaled by log2e/sqrt(hd));
// V: bf16 [bh][64][t] (head-transposed). Block = 128 thr (4 waves) sharing one
// (b,h); each wave owns 16 query rows. 32-key blocks staged in LDS via async
// b128 copies (no VGPR round-trip).
// ---------------------------------------------------------------------------
__global__ __launch_bounds__(128, 2)
void flash_attn_kernel(const bf16* __restrict__ Q, const bf16* __restrict__ K,
                       const bf16* __restrict__ Vt, bf16* __restrict__ O)
{
    __shared__ bf16 Ks[32 * 72];       // [key][feat]; 144B rows (16B aligned)
    __shared__ bf16 Vs[64 * 40];       // [dim][key];   80B rows (16B aligned)
    __shared__ bf16 Ps[4][16 * 36];    // per-wave P staging for C->A transpose

    const int tid  = threadIdx.x;
    const int lane = tid & 31;
    const int wid  = tid >> 5;
    const int bh   = blockIdx.x >> 5;          // 32 q-groups per (b,h)
    const int qg   = blockIdx.x & 31;
    const int q0   = qg * 64 + wid * 16;
    const int b    = bh >> 4, h = bh & 15;

    // Q tile as two A-frags (features 0-31 / 32-63), loaded once.
    const bf16* Qrow = Q + ((size_t)bh * T_ + q0 + (lane & 15)) * HD_;
    v16bf aq0, aq1;
    {
        union { v16bf bv; v8u u; } f0, f1;
        #pragma unroll
        for (int v = 0; v < 8; ++v) {
            int k = frag_k(v, lane);
            f0.u[v] = *(const unsigned int*)(Qrow + k);
            f1.u[v] = *(const unsigned int*)(Qrow + k + 32);
        }
        aq0 = f0.bv; aq1 = f1.bv;
    }

    v8f o[4];
    #pragma unroll
    for (int nc = 0; nc < 4; ++nc) o[nc] = v8f_zero();
    float mrow[8], lrow[8];
    #pragma unroll
    for (int r = 0; r < 8; ++r) { mrow[r] = -1e30f; lrow[r] = 0.0f; }

    const bf16* Kbh = K  + (size_t)bh * T_ * HD_;
    const bf16* Vbh = Vt + (size_t)bh * HD_ * T_;

    for (int s0 = 0; s0 < T_; s0 += 32) {
        __syncthreads();
        // K block: 32 keys x 64 feats = 256 x 16B chunks, 2 per thread.
        #pragma unroll
        for (int c = tid; c < 256; c += 128) {
            int key = c >> 3, ch = c & 7;
            async_b128(Kbh + (size_t)(s0 + key) * HD_ + ch * 8, &Ks[key * 72 + ch * 8]);
        }
        // V block (already transposed): 64 dims x 32 keys = 256 x 16B chunks.
        #pragma unroll
        for (int c = tid; c < 256; c += 128) {
            int d = c >> 2, ch = c & 3;
            async_b128(Vbh + (size_t)d * T_ + s0 + ch * 8, &Vs[d * 40 + ch * 8]);
        }
        wait_async();
        __syncthreads();

        // S(16x32) = Q(16x64) * K^T : 2 key subtiles x 2 K-depth WMMAs
        v8f s[2];
        #pragma unroll
        for (int st = 0; st < 2; ++st) {
            const bf16* krow = &Ks[(st * 16 + (lane & 15)) * 72];
            union { v16bf bv; v8u u; } g0, g1;
            #pragma unroll
            for (int v = 0; v < 8; ++v) {
                int k = frag_k(v, lane);
                g0.u[v] = *(const unsigned int*)(krow + k);
                g1.u[v] = *(const unsigned int*)(krow + k + 32);
            }
            v8f acc = v8f_zero();
            acc = __builtin_amdgcn_wmma_f32_16x16x32_bf16(
                false, aq0, false, g0.bv, (short)0, acc, false, false);
            acc = __builtin_amdgcn_wmma_f32_16x16x32_bf16(
                false, aq1, false, g1.bv, (short)0, acc, false, false);
            s[st] = acc;
        }

        // Online softmax in exp2 domain (scale folded into Q at projection).
        // Each row lives in one 16-lane half -> xor-reduce within the half.
        #pragma unroll
        for (int r = 0; r < 8; ++r) {
            float v0 = s[0][r];
            float v1 = s[1][r];
            float mx = fmaxf(v0, v1);
            #pragma unroll
            for (int d = 1; d < 16; d <<= 1) mx = fmaxf(mx, __shfl_xor(mx, d, 32));
            float mnew  = fmaxf(mrow[r], mx);
            float alpha = exp2f(mrow[r] - mnew);
            float p0 = exp2f(v0 - mnew);
            float p1 = exp2f(v1 - mnew);
            float ps = p0 + p1;
            #pragma unroll
            for (int d = 1; d < 16; d <<= 1) ps += __shfl_xor(ps, d, 32);
            lrow[r] = lrow[r] * alpha + ps;
            mrow[r] = mnew;
            #pragma unroll
            for (int nc = 0; nc < 4; ++nc) o[nc][r] *= alpha;

            int m = r + ((lane & 16) ? 8 : 0);
            Ps[wid][m * 36 + (lane & 15)]      = (bf16)p0;
            Ps[wid][m * 36 + 16 + (lane & 15)] = (bf16)p1;
        }
        __syncthreads();   // uniform trip counts; completes P C->A transpose

        // O(16x64) += P(16x32) * V(32x64): 4 dim subtiles
        v16bf pf = load_frag_row(&Ps[wid][(lane & 15) * 36], lane);
        #pragma unroll
        for (int nc = 0; nc < 4; ++nc) {
            const bf16* vrow = &Vs[(nc * 16 + (lane & 15)) * 40];
            v16bf vb = load_frag_row(vrow, lane);
            o[nc] = __builtin_amdgcn_wmma_f32_16x16x32_bf16(
                false, pf, false, vb, (short)0, o[nc], false, false);
        }
    }

    // Normalize and store attn-out as flat (b, t, d) bf16 for the Wo GEMM.
    #pragma unroll
    for (int r = 0; r < 8; ++r) {
        float inv = 1.0f / lrow[r];
        int m = r + ((lane & 16) ? 8 : 0);
        int t = q0 + m;
        #pragma unroll
        for (int nc = 0; nc < 4; ++nc) {
            int dcol = h * HD_ + nc * 16 + (lane & 15);
            O[((size_t)b * T_ + t) * D_ + dcol] = (bf16)(o[nc][r] * inv);
        }
    }
}

// ---------------------------------------------------------------------------
extern "C" void kernel_launch(void* const* d_in, const int* in_sizes, int n_in,
                              void* d_out, int out_size, void* d_ws, size_t ws_size,
                              hipStream_t stream)
{
    (void)in_sizes; (void)n_in; (void)out_size; (void)ws_size;
    const float* k_in = (const float*)d_in[0];
    const float* q_in = (const float*)d_in[1];
    const float* v_in = (const float*)d_in[2];
    const float* Wk   = (const float*)d_in[3];
    const float* Wq   = (const float*)d_in[4];
    const float* Wv   = (const float*)d_in[5];
    const float* Wo   = (const float*)d_in[6];
    const float* bo   = (const float*)d_in[7];

    const size_t NQKV = (size_t)B_ * H_ * T_ * HD_;   // 4,194,304 elems
    bf16* Qp = (bf16*)d_ws;                            // [b,h,t,64], pre-scaled
    bf16* Kp = Qp + NQKV;                              // [b,h,t,64]
    bf16* Vp = Kp + NQKV;                              // [b,h,64,t] transposed
    bf16* Ao = Vp + NQKV;                              // flat (b,t,d)

    // Softmax scale 1/sqrt(64) and log2(e) folded into Q so the flash kernel
    // runs its online softmax purely in the exp2 domain.
    const float qscale = 0.125f * 1.4426950408889634f;

    dim3 ggrid(D_ / 64, BT_ / 64);                     // (16, 64)
    gemm_wmma_kernel<false, 0><<<ggrid, 128, 0, stream>>>(q_in, Wq, nullptr, Qp, qscale);
    gemm_wmma_kernel<false, 0><<<ggrid, 128, 0, stream>>>(k_in, Wk, nullptr, Kp, 1.0f);
    gemm_wmma_kernel<false, 1><<<ggrid, 128, 0, stream>>>(v_in, Wv, nullptr, Vp, 1.0f);

    flash_attn_kernel<<<B_ * H_ * (T_ / 64), 128, 0, stream>>>(Qp, Kp, Vp, Ao);

    gemm_wmma_kernel<true, 2><<<ggrid, 128, 0, stream>>>(Ao, Wo, bo, (float*)d_out, 1.0f);
}
